// GATLayer_89807766159586
// MI455X (gfx1250) — matrix operand order repaired and checked by reading, not compiled, not run on hardware
//
#include <hip/hip_runtime.h>
#include <hip/hip_bf16.h>
#include <math.h>

// ---------------- problem constants (match reference) ----------------
#define N_NODES 8192
#define NIN     512
#define NOUT    256
#define NEDGE   262144
#define NEG_SLOPE 0.2f
#define BN_EPS    1e-5f
#define ADJ_WORDS 256          // 8192 cols / 32 bits
#define CHUNK_WORDS 64         // 2048 columns per chunk in aggregation

typedef __attribute__((ext_vector_type(16))) _Float16 v16h;
typedef __attribute__((ext_vector_type(8)))  float    v8f;
typedef unsigned int u32;

#define WMMA_F16(A, B, C) __builtin_amdgcn_wmma_f32_16x16x32_f16( \
    false, (A), false, (B), (short)0, (C), false, false)

// ---------------- K0: zero the adjacency bitmask ----------------
__global__ void zero_kernel(u32* __restrict__ p, int n) {
    int i = blockIdx.x * blockDim.x + threadIdx.x;
    if (i < n) p[i] = 0u;
}

// ---------------- K1: build adjacency bitmask (idempotent atomics) ----------------
__global__ void adj_kernel(const int* __restrict__ edge_index, u32* __restrict__ adj) {
    int k = blockIdx.x * blockDim.x + threadIdx.x;
    if (k >= NEDGE) return;
    int r = edge_index[k];
    int c = edge_index[NEDGE + k];
    atomicOr(adj + r * ADJ_WORDS + (c >> 5), 1u << (c & 31));
}

// ---------------- K2: h = X @ W via WMMA f32_16x16x32_f16 ----------------
// One wave computes a 32x32 output macro-tile (2x2 WMMA tiles): 4 WMMAs per
// K-step with 2 A-frags + 2 B-frags -> 2x register reuse on both operands.
// 2048 waves total = 256 blocks x 8 waves.
__global__ void gemm_h_kernel(const float* __restrict__ X,
                              const float* __restrict__ W,
                              float* __restrict__ h) {
    const int wave = threadIdx.x >> 5;
    const int lane = threadIdx.x & 31;
    const int gw   = blockIdx.x * 8 + wave;     // 0..2047
    const int cs   = gw & 7;                    // col super-tile (32 cols)
    const int rs   = gw >> 3;                   // row super-tile (32 rows)
    const int g    = lane >> 4;                 // lane group 0/1
    const int l16  = lane & 15;
    const int row0 = rs * 32 + l16;             // A row, tile row 0
    const int row1 = row0 + 16;                 // A row, tile row 1
    const int col0 = cs * 32 + l16;             // B/D col, tile col 0
    const int col1 = col0 + 16;

    v8f acc00 = {}, acc01 = {}, acc10 = {}, acc11 = {};

    for (int kt = 0; kt < NIN; kt += 32) {
        // A fragments (16x32 f16): lane holds K run [kt+16g,+15]
        const float* xp0 = X + (size_t)row0 * NIN + kt + 16 * g;
        const float* xp1 = X + (size_t)row1 * NIN + kt + 16 * g;
        __builtin_prefetch(xp0 + 32, 0, 1);     // next K-step of the A stream
        __builtin_prefetch(xp1 + 32, 0, 1);
        v16h a0, a1;
#pragma unroll
        for (int e = 0; e < 8; ++e) {
            a0[e]     = (_Float16)xp0[e];
            a0[8 + e] = (_Float16)xp0[8 + e];
            a1[e]     = (_Float16)xp1[e];
            a1[8 + e] = (_Float16)xp1[8 + e];
        }
        // B fragments (32x16 f16): lane = column, K = kt + 16g + e
        const float* wp = W + (size_t)(kt + 16 * g) * NOUT;
        v16h b0, b1;
#pragma unroll
        for (int e = 0; e < 16; ++e) {
            b0[e] = (_Float16)wp[e * NOUT + col0];
            b1[e] = (_Float16)wp[e * NOUT + col1];
        }

        acc00 = WMMA_F16(a0, b0, acc00);
        acc01 = WMMA_F16(a0, b1, acc01);
        acc10 = WMMA_F16(a1, b0, acc10);
        acc11 = WMMA_F16(a1, b1, acc11);
    }
    // D layout: VGPR r -> row (tileRowBase + r + 8g), col = lane's column
#pragma unroll
    for (int r = 0; r < 8; ++r) {
        const int rr0 = rs * 32 + r + 8 * g;
        const int rr1 = rr0 + 16;
        h[(size_t)rr0 * NOUT + col0] = acc00[r];
        h[(size_t)rr0 * NOUT + col1] = acc01[r];
        h[(size_t)rr1 * NOUT + col0] = acc10[r];
        h[(size_t)rr1 * NOUT + col1] = acc11[r];
    }
}

// ---------------- K3: s1 = h@a1, s2 = h@a2 (one wave per row) ----------------
__global__ void s1s2_kernel(const float* __restrict__ h, const float* __restrict__ a,
                            float* __restrict__ s1, float* __restrict__ s2) {
    int wave = threadIdx.x >> 5, lane = threadIdx.x & 31;
    int row = blockIdx.x * 8 + wave;
    const float* hr = h + (size_t)row * NOUT;
    float p1 = 0.f, p2 = 0.f;
#pragma unroll
    for (int e = 0; e < 8; ++e) {
        int c = lane + e * 32;               // coalesced
        float hv = hr[c];
        p1 += hv * a[c];
        p2 += hv * a[NOUT + c];
    }
#pragma unroll
    for (int off = 16; off; off >>= 1) {
        p1 += __shfl_xor(p1, off, 32);
        p2 += __shfl_xor(p2, off, 32);
    }
    if (lane == 0) { s1[row] = p1; s2[row] = p2; }
}

__device__ __forceinline__ float lrelu(float x) {
    return x > 0.f ? x : NEG_SLOPE * x;
}

// ---------------- K4: per-row softmax stats over edge set (one wave per row) ----------------
__global__ void rowstats_kernel(const u32* __restrict__ adj,
                                const float* __restrict__ s1, const float* __restrict__ s2,
                                float* __restrict__ rowmax, float* __restrict__ rowsum,
                                int* __restrict__ rowcnt) {
    int wave = threadIdx.x >> 5, lane = threadIdx.x & 31;
    int row = blockIdx.x * 8 + wave;
    const u32* arow = adj + (size_t)row * ADJ_WORDS;
    float s1v = s1[row];
    float m = -INFINITY; int cnt = 0;
    for (int wi = lane; wi < ADJ_WORDS; wi += 32) {
        u32 bits = arow[wi];
        cnt += __popc(bits);
        while (bits) {
            int b = __ffs(bits) - 1; bits &= bits - 1;
            int j = wi * 32 + b;
            m = fmaxf(m, lrelu(s1v + s2[j]));
        }
    }
#pragma unroll
    for (int off = 16; off; off >>= 1) {
        m = fmaxf(m, __shfl_xor(m, off, 32));
        cnt += __shfl_xor(cnt, off, 32);
    }
    float sum = 0.f;
    for (int wi = lane; wi < ADJ_WORDS; wi += 32) {
        u32 bits = arow[wi];
        while (bits) {
            int b = __ffs(bits) - 1; bits &= bits - 1;
            int j = wi * 32 + b;
            sum += __expf(lrelu(s1v + s2[j]) - m);
        }
    }
#pragma unroll
    for (int off = 16; off; off >>= 1) sum += __shfl_xor(sum, off, 32);
    if (lane == 0) { rowmax[row] = m; rowsum[row] = sum; rowcnt[row] = cnt; }
}

// ---------------- K5: out_raw = att @ h (sparse, deterministic prefix-scan list) ----------------
__global__ void aggregate_kernel(const u32* __restrict__ adj, const float* __restrict__ h,
                                 const float* __restrict__ s1, const float* __restrict__ s2,
                                 const float* __restrict__ rowmax, const float* __restrict__ rowsum,
                                 const int* __restrict__ rowcnt, float* __restrict__ out_raw) {
    __shared__ int   s_scan[CHUNK_WORDS];
    __shared__ int   s_j[CHUNK_WORDS * 32];
    __shared__ float s_w[CHUNK_WORDS * 32];

    const int row = blockIdx.x;
    const int t   = threadIdx.x;                 // 0..255 == output column
    const u32* arow = adj + (size_t)row * ADJ_WORDS;
    const float s1v = s1[row];
    const float mx  = rowmax[row];
    const float sm  = rowsum[row];
    const bool  empty = (rowcnt[row] == 0);      // reference: uniform softmax row
    const float invN = 1.0f / (float)N_NODES;

    float acc = 0.f;
    for (int chunk = 0; chunk < ADJ_WORDS / CHUNK_WORDS; ++chunk) {
        u32 bits = 0; int mycnt = 0;
        if (t < CHUNK_WORDS) {
            bits = empty ? 0xFFFFFFFFu : arow[chunk * CHUNK_WORDS + t];
            mycnt = __popc(bits);
            s_scan[t] = mycnt;
        }
        __syncthreads();
        // Hillis-Steele inclusive scan over CHUNK_WORDS counts (deterministic ordering)
        for (int off = 1; off < CHUNK_WORDS; off <<= 1) {
            int v = 0;
            if (t < CHUNK_WORDS) { v = s_scan[t]; if (t >= off) v += s_scan[t - off]; }
            __syncthreads();
            if (t < CHUNK_WORDS) s_scan[t] = v;
            __syncthreads();
        }
        if (t < CHUNK_WORDS) {
            int pos = s_scan[t] - mycnt;         // exclusive offset
            int base = (chunk * CHUNK_WORDS + t) * 32;
            u32 bb = bits;
            while (bb) {
                int b = __ffs(bb) - 1; bb &= bb - 1;
                int j = base + b;
                float w;
                if (empty) w = invN;
                else       w = __expf(lrelu(s1v + s2[j]) - mx) / sm;
                s_j[pos] = j; s_w[pos] = w; ++pos;
            }
        }
        __syncthreads();
        int total = s_scan[CHUNK_WORDS - 1];
        for (int idx = 0; idx < total; ++idx)
            acc += s_w[idx] * h[(size_t)s_j[idx] * NOUT + t];   // coalesced across t
        __syncthreads();
    }
    out_raw[(size_t)row * NOUT + t] = acc;
}

// ---------------- K6: alpha = att[row, col] per edge ----------------
__global__ void alpha_kernel(const int* __restrict__ edge_index,
                             const float* __restrict__ s1, const float* __restrict__ s2,
                             const float* __restrict__ rowmax, const float* __restrict__ rowsum,
                             const int* __restrict__ rowcnt, float* __restrict__ alpha) {
    int k = blockIdx.x * blockDim.x + threadIdx.x;
    if (k >= NEDGE) return;
    int r = edge_index[k];
    int c = edge_index[NEDGE + k];
    float out;
    if (rowcnt[r] == 0) out = 1.0f / (float)N_NODES;
    else out = __expf(lrelu(s1[r] + s2[c]) - rowmax[r]) / rowsum[r];
    alpha[k] = out;
}

// ---------------- K7a: BN partial column sums (deterministic, no atomics) ----------------
__global__ void bnpart_kernel(const float* __restrict__ out_raw, float* __restrict__ partial) {
    int c = threadIdx.x, b = blockIdx.x;        // 64 blocks x 128 rows each
    float s = 0.f, q = 0.f;
    for (int r = b * 128; r < (b + 1) * 128; ++r) {
        float v = out_raw[(size_t)r * NOUT + c];
        s += v; q += v * v;
    }
    partial[b * NOUT + c] = s;
    partial[64 * NOUT + b * NOUT + c] = q;
}

// ---------------- K7b: finalize mean/var ----------------
__global__ void bnstats_kernel(const float* __restrict__ partial, float* __restrict__ meanvar) {
    int c = threadIdx.x;
    float s = 0.f, q = 0.f;
    for (int b = 0; b < 64; ++b) {
        s += partial[b * NOUT + c];
        q += partial[64 * NOUT + b * NOUT + c];
    }
    float mean = s / (float)N_NODES;
    float var  = q / (float)N_NODES - mean * mean;
    meanvar[c] = mean;
    meanvar[NOUT + c] = var;
}

// ---------------- K8: BN apply + bias + ELU ----------------
__global__ void bnapply_kernel(const float* __restrict__ out_raw, const float* __restrict__ meanvar,
                               const float* __restrict__ gamma, const float* __restrict__ beta,
                               const float* __restrict__ bias, float* __restrict__ out) {
    int idx = blockIdx.x * blockDim.x + threadIdx.x;   // over N*NOUT
    int c = idx & (NOUT - 1);
    float mean = meanvar[c], var = meanvar[NOUT + c];
    float v = (out_raw[idx] - mean) * rsqrtf(var + BN_EPS) * gamma[c] + beta[c] + bias[c];
    out[idx] = v > 0.f ? v : (__expf(v) - 1.0f);
}

// ---------------- host side ----------------
extern "C" void kernel_launch(void* const* d_in, const int* in_sizes, int n_in,
                              void* d_out, int out_size, void* d_ws, size_t ws_size,
                              hipStream_t stream) {
    (void)in_sizes; (void)n_in; (void)out_size; (void)ws_size;
    const float* X     = (const float*)d_in[0];   // 8192 x 512
    const float* W     = (const float*)d_in[1];   // 512 x 256
    const float* a     = (const float*)d_in[2];   // 512 x 1
    const float* gamma = (const float*)d_in[3];   // 256
    const float* beta  = (const float*)d_in[4];   // 256
    const float* bias  = (const float*)d_in[5];   // 256
    const int*   EI    = (const int*)d_in[6];     // 2 x 262144

    float* out   = (float*)d_out;                 // N*NOUT floats
    float* alpha = out + (size_t)N_NODES * NOUT;  // then E floats

    // workspace layout (all 4-byte elements)
    float* ws      = (float*)d_ws;
    float* h       = ws;                                    // 2,097,152
    u32*   adj     = (u32*)(ws + 2097152);                  // 2,097,152 words
    float* out_raw = ws + 2 * 2097152;                      // 2,097,152
    float* s1      = ws + 3 * 2097152;                      // 8192
    float* s2      = s1 + N_NODES;
    float* rowmax  = s2 + N_NODES;
    float* rowsum  = rowmax + N_NODES;
    int*   rowcnt  = (int*)(rowsum + N_NODES);
    float* partial = (float*)(rowcnt + N_NODES);            // 2*64*256
    float* meanvar = partial + 2 * 64 * NOUT;               // 2*256

    const int adj_words = N_NODES * ADJ_WORDS;              // 2,097,152

    zero_kernel<<<(adj_words + 255) / 256, 256, 0, stream>>>(adj, adj_words);
    adj_kernel<<<(NEDGE + 255) / 256, 256, 0, stream>>>(EI, adj);
    // 32x32 macro-tiles: (8192/32)*(256/32) = 2048 waves = 256 blocks of 8 waves
    gemm_h_kernel<<<256, 256, 0, stream>>>(X, W, h);
    s1s2_kernel<<<N_NODES / 8, 256, 0, stream>>>(h, a, s1, s2);
    rowstats_kernel<<<N_NODES / 8, 256, 0, stream>>>(adj, s1, s2, rowmax, rowsum, rowcnt);
    aggregate_kernel<<<N_NODES, NOUT, 0, stream>>>(adj, h, s1, s2, rowmax, rowsum, rowcnt, out_raw);
    alpha_kernel<<<(NEDGE + 255) / 256, 256, 0, stream>>>(EI, s1, s2, rowmax, rowsum, rowcnt, alpha);
    bnpart_kernel<<<64, NOUT, 0, stream>>>(out_raw, partial);
    bnstats_kernel<<<1, NOUT, 0, stream>>>(partial, meanvar);
    bnapply_kernel<<<(N_NODES * NOUT) / 256, 256, 0, stream>>>(out_raw, meanvar, gamma, beta, bias, out);
}
